// GlobalFeatureExtractor_4466765988307
// MI455X (gfx1250) — matrix operand order repaired and checked by reading, not compiled, not run on hardware
//
#include <hip/hip_runtime.h>
#include <hip/hip_bf16.h>

// ---------------------------------------------------------------------------
// GlobalFeatureExtractor for MI455X (gfx1250), wave32 + WMMA bf16 16x16x32.
//
// Pipeline (all on `stream`, scratch in d_ws):
//  K0  zero stats accumulators
//  Kp  pre-pack weight matrices into WMMA A-fragment order (bf16)
//  K1  LN1:        x -> HN
//  K2  dw 3x3:     HN -> HH                (zero 'SAME' pad)
//  K3  WMMA qkv:   HH x {Wq,Wk,Wv} -> Q,K,V   (fused, one B-frag load)
//  K4  stats:      Q,K,V -> sumsq[3][B][64], Gram[B][8][8][8] (atomics)
//  K5  attn:       softmax + fold proj: Wcomb[b] = proj * BlockDiag(attn/|v|)
//                  -> bf16 A-fragments per batch
//  K6  WMMA:       d_out = x + Wcomb[b] * V      (attention+proj+residual)
//  K7  LN2:        d_out -> M0
//  K8  WMMA:       M1(256ch) = mlp_in_w * M0
//  K9  fused grouped dwconv (reflect pad) + SiLU gate + BN -> S
//  K11 WMMA:       d_out += out_w * S            (final residual, in-place)
// ---------------------------------------------------------------------------

typedef __attribute__((ext_vector_type(16))) __bf16 v16bf;
typedef __attribute__((ext_vector_type(8)))  float  v8f;

union BF16x16 { unsigned short s[16]; v16bf v; };

static __device__ __forceinline__ unsigned short f2bf(float f) {
  union { float f; unsigned int u; } x; x.f = f;
  unsigned int u = x.u;
  return (unsigned short)((u + 0x7FFFu + ((u >> 16) & 1u)) >> 16);
}

constexpr int    BATCH = 4;
constexpr int    CH    = 64;
constexpr int    IMG   = 256;
constexpr size_t NPOS  = (size_t)IMG * IMG;          // 65536
constexpr size_t T_EL  = (size_t)BATCH * CH * NPOS;  // 16,777,216 elements
constexpr size_t T_B   = T_EL * 4;                   // 67,108,864 bytes

// d_ws layout (bytes)
constexpr size_t HN_OFF    = 0;
constexpr size_t HH_OFF    = 1 * T_B;
constexpr size_t Q_OFF     = 2 * T_B;
constexpr size_t K_OFF     = 3 * T_B;
constexpr size_t V_OFF     = 4 * T_B;
constexpr size_t M1_OFF    = 1 * T_B;        // [B][256][N], reuses HH..V after K6
constexpr size_t SSQ_OFF   = 5 * T_B;                 // 768 f32 ([3][B][64])
constexpr size_t GRAM_OFF  = SSQ_OFF  + 4096;         // 2048 f32
constexpr size_t WCOMB_OFF = GRAM_OFF + 8192;         // 16384 f32
constexpr size_t FRQ_OFF   = WCOMB_OFF + 65536;       // 3*4096 bf16 (q,k,v frags)
constexpr size_t FRM_OFF   = FRQ_OFF + 24576;         // 16384 bf16 (mlp_in frags)
constexpr size_t FRO_OFF   = FRM_OFF + 32768;         // 4096 bf16 (out_w frags)
constexpr size_t FRW_OFF   = FRO_OFF + 8192;          // 16384 bf16 (Wcomb frags)

// A-fragment k index for 16-bit 16x32 A (ISA 05_wmma §7.12.2)
static __device__ __forceinline__ int a_kmap(int i, int half, int slot) {
  return ((i < 4) ? (i * 2) : (16 + (i - 4) * 2)) + half * 8 + slot;
}

// -------------------------------- K0: zero ---------------------------------
__global__ void k_zero(float* __restrict__ p, int n) {
  int i = blockIdx.x * blockDim.x + threadIdx.x;
  if (i < n) p[i] = 0.0f;
}

// ------------------------- Kp: pack weight fragments -----------------------
// dst layout: [mt][ks][lane][i*2+slot] halves; one 16x Cin=64 tile set.
__global__ void k_prep(const float* __restrict__ W /*[Cout][64]*/, int MT,
                       unsigned short* __restrict__ dst) {
  int idx = blockIdx.x * blockDim.x + threadIdx.x;
  if (idx >= MT * 1024) return;
  int mt   = idx >> 10;
  int r2   = idx & 1023;
  int ks   = r2 >> 9;
  int r3   = r2 & 511;
  int lane = r3 >> 4;
  int t    = r3 & 15;
  int half = lane >> 4;
  int m    = mt * 16 + (lane & 15);
  int k    = ks * 32 + a_kmap(t >> 1, half, t & 1);
  dst[idx] = f2bf(W[m * 64 + k]);
}

// ------------------------------ K1/K7: LayerNorm ---------------------------
__global__ __launch_bounds__(256) void k_ln(const float* __restrict__ x,
                                            const float* __restrict__ w,
                                            const float* __restrict__ bia,
                                            float* __restrict__ out) {
  size_t p = (size_t)blockIdx.x * blockDim.x + threadIdx.x; // B*N threads
  int b = (int)(p >> 16);
  size_t n = p & 65535;
  const float* xb = x + ((size_t)b * CH) * NPOS + n;
  float v[CH];
  float mu = 0.f;
#pragma unroll
  for (int c = 0; c < CH; ++c) { v[c] = xb[(size_t)c * NPOS]; mu += v[c]; }
  mu *= (1.0f / CH);
  float var = 0.f;
#pragma unroll
  for (int c = 0; c < CH; ++c) { float d = v[c] - mu; var += d * d; }
  var *= (1.0f / CH);
  float inv = rsqrtf(var + 1e-5f);
  float* ob = out + ((size_t)b * CH) * NPOS + n;
#pragma unroll
  for (int c = 0; c < CH; ++c)
    ob[(size_t)c * NPOS] = (v[c] - mu) * inv * w[c] + bia[c];
}

// --------------------------- K2: depthwise 3x3 (SAME) ----------------------
__global__ __launch_bounds__(256) void k_dw(const float* __restrict__ in,
                                            const float* __restrict__ w,
                                            float* __restrict__ out) {
  size_t idx = (size_t)blockIdx.x * blockDim.x + threadIdx.x; // B*64*N
  int n = (int)(idx & 65535);
  int c = (int)((idx >> 16) & 63);
  int b = (int)(idx >> 22);
  int y = n >> 8, xx = n & 255;
  const float* ip = in + (((size_t)b * CH + c) << 16);
  const float* wc = w + c * 9;
  float acc = 0.f;
#pragma unroll
  for (int dy = -1; dy <= 1; ++dy) {
    int yy = y + dy;
    if (yy < 0 || yy >= IMG) continue;
#pragma unroll
    for (int dx = -1; dx <= 1; ++dx) {
      int x2 = xx + dx;
      if (x2 < 0 || x2 >= IMG) continue;
      acc += wc[(dy + 1) * 3 + (dx + 1)] * ip[yy * IMG + x2];
    }
  }
  out[idx] = acc;
}

// --------------------- generic WMMA 1x1-conv (K=64, bf16) ------------------
template <int NOUT, int MT, bool RESID, bool BATCHW>
__global__ __launch_bounds__(256) void k_wmma(
    const float* __restrict__ in,           // [B][64][N] fp32
    const unsigned short* __restrict__ wf,  // packed bf16 A-fragments
    float* __restrict__ out,                // [B][MT*16][N] (+ w*ostride)
    const float* __restrict__ res,          // residual or nullptr
    size_t ostride) {
  constexpr int TPW = 8;
  constexpr int NG  = (int)(NPOS / 16) / TPW;  // 512
  const int lane = threadIdx.x & 31;
  const int wid  = blockIdx.x * (blockDim.x >> 5) + (threadIdx.x >> 5);
  const int mt   = wid % MT;
  const int ng   = (wid / MT) % NG;
  const int b    = wid / (MT * NG);
  const int half = lane >> 4;
  const int nlo  = lane & 15;

  // A fragments: contiguous 16 halves per lane (packed by k_prep / k_attn)
  v16bf afr[NOUT][2];
  const size_t wbase = BATCHW ? (size_t)b * ((size_t)MT * 2 * 512) : 0;
#pragma unroll
  for (int w = 0; w < NOUT; ++w)
#pragma unroll
    for (int ks = 0; ks < 2; ++ks)
      afr[w][ks] = *reinterpret_cast<const v16bf*>(
          wf + wbase + (((size_t)w * MT + mt) * 2 + ks) * 512 +
          (size_t)lane * 16);

  const float* inb = in + ((size_t)b * CH) * NPOS;

  for (int t = 0; t < TPW; ++t) {
    const int nbase = (ng * TPW + t) * 16;
    const int n = nbase + nlo;
    if (t + 1 < TPW)  // lowers to global_prefetch_b8
      __builtin_prefetch(inb + (size_t)(half * 16) * NPOS + n + 16, 0, 1);
    BF16x16 bfr[2];
#pragma unroll
    for (int ks = 0; ks < 2; ++ks)
#pragma unroll
      for (int j = 0; j < 8; ++j)
#pragma unroll
        for (int slot = 0; slot < 2; ++slot) {
          int kk = ks * 32 + half * 8 + (j >> 2) * 16 + (j & 3) * 2 + slot;
          bfr[ks].s[j * 2 + slot] = f2bf(inb[(size_t)kk * NPOS + n]);
        }
#pragma unroll
    for (int w = 0; w < NOUT; ++w) {
      v8f acc = {0.f, 0.f, 0.f, 0.f, 0.f, 0.f, 0.f, 0.f};
      acc = __builtin_amdgcn_wmma_f32_16x16x32_bf16(
          false, afr[w][0], false, bfr[0].v, (short)0, acc, false, false);
      acc = __builtin_amdgcn_wmma_f32_16x16x32_bf16(
          false, afr[w][1], false, bfr[1].v, (short)0, acc, false, false);
#pragma unroll
      for (int r = 0; r < 8; ++r) {
        int m = mt * 16 + r + 8 * half;
        size_t bi = ((size_t)b * (MT * 16) + m) * NPOS + n;
        float vl = acc[r];
        if (RESID) vl += res[bi];
        out[bi + (size_t)w * ostride] = vl;
      }
    }
  }
}

// ---------------- K4: per-channel sumsq + per-head Gram (q·kᵀ) -------------
__global__ __launch_bounds__(256) void k_stats(const float* __restrict__ q,
                                               const float* __restrict__ k,
                                               const float* __restrict__ v,
                                               float* __restrict__ ssq,   // [3][B][64]
                                               float* __restrict__ gram) { // [B][8][8][8]
  const int bh = blockIdx.y;             // b*8 + h
  const int b = bh >> 3, h = bh & 7;
  const int tid = threadIdx.x;
  const size_t base = ((size_t)b * CH + h * 8) * NPOS;
  const float* qb = q + base;
  const float* kb = k + base;
  const float* vb = v + base;
  float g[64], sq[8], sk[8], sv[8];
#pragma unroll
  for (int i = 0; i < 64; ++i) g[i] = 0.f;
#pragma unroll
  for (int i = 0; i < 8; ++i) { sq[i] = 0.f; sk[i] = 0.f; sv[i] = 0.f; }
  for (int it = 0; it < 8; ++it) {
    size_t n = (size_t)blockIdx.x * 2048 + it * 256 + tid;
    float qv[8], kv[8], vv[8];
#pragma unroll
    for (int d = 0; d < 8; ++d) {
      qv[d] = qb[(size_t)d * NPOS + n];
      kv[d] = kb[(size_t)d * NPOS + n];
      vv[d] = vb[(size_t)d * NPOS + n];
    }
#pragma unroll
    for (int d = 0; d < 8; ++d) {
      sq[d] += qv[d] * qv[d];
      sk[d] += kv[d] * kv[d];
      sv[d] += vv[d] * vv[d];
#pragma unroll
      for (int e = 0; e < 8; ++e) g[d * 8 + e] += qv[d] * kv[e];
    }
  }
  __shared__ float red[88];
  if (tid < 88) red[tid] = 0.f;
  __syncthreads();
#pragma unroll
  for (int i = 0; i < 64; ++i) atomicAdd(&red[i], g[i]);
#pragma unroll
  for (int d = 0; d < 8; ++d) {
    atomicAdd(&red[64 + d], sq[d]);
    atomicAdd(&red[72 + d], sk[d]);
    atomicAdd(&red[80 + d], sv[d]);
  }
  __syncthreads();
  if (tid < 64)       atomicAdd(&gram[(size_t)bh * 64 + tid], red[tid]);
  else if (tid < 72)  atomicAdd(&ssq[0 * 256 + b * 64 + h * 8 + (tid - 64)], red[tid]);
  else if (tid < 80)  atomicAdd(&ssq[1 * 256 + b * 64 + h * 8 + (tid - 72)], red[tid]);
  else if (tid < 88)  atomicAdd(&ssq[2 * 256 + b * 64 + h * 8 + (tid - 80)], red[tid]);
}

// ------- K5: softmax + fold proj into per-batch 64x64 Wcomb, pack frag -----
__global__ __launch_bounds__(256) void k_attn(const float* __restrict__ ssq,
                                              const float* __restrict__ gram,
                                              const float* __restrict__ proj,
                                              float* __restrict__ wcomb,       // [4][64][64]
                                              unsigned short* __restrict__ frag) {
  __shared__ float A2[BATCH][8][8][8];
  const int t = threadIdx.x;              // 256 = (b,h,d)
  const int b = t >> 6, h = (t >> 3) & 7, d = t & 7;
  const float scale = 0.35355339059327373f;  // 8^-0.5
  float invq = 1.0f / fmaxf(sqrtf(ssq[0 * 256 + b * 64 + h * 8 + d]), 1e-12f);
  float row[8];
#pragma unroll
  for (int e = 0; e < 8; ++e) {
    float invk = 1.0f / fmaxf(sqrtf(ssq[1 * 256 + b * 64 + h * 8 + e]), 1e-12f);
    row[e] = gram[((size_t)(b * 8 + h)) * 64 + d * 8 + e] * scale * invq * invk;
  }
  float mx = row[0];
#pragma unroll
  for (int e = 1; e < 8; ++e) mx = fmaxf(mx, row[e]);
  float s = 0.f;
#pragma unroll
  for (int e = 0; e < 8; ++e) { row[e] = __expf(row[e] - mx); s += row[e]; }
  float invs = 1.0f / s;
#pragma unroll
  for (int e = 0; e < 8; ++e) {
    float invv = 1.0f / fmaxf(sqrtf(ssq[2 * 256 + b * 64 + h * 8 + e]), 1e-12f);
    A2[b][h][d][e] = row[e] * invs * invv;
  }
  __syncthreads();
  // Wcomb[b][o][c] = sum_d proj[o][h*8+d] * A2[b][h][d][e],  c = h*8+e
  for (int idx = t; idx < BATCH * 64 * 64; idx += 256) {
    int bb = idx >> 12, o = (idx >> 6) & 63, c = idx & 63;
    int hh = c >> 3, e = c & 7;
    float acc = 0.f;
#pragma unroll
    for (int dd = 0; dd < 8; ++dd) acc += proj[o * 64 + hh * 8 + dd] * A2[bb][hh][dd][e];
    wcomb[idx] = acc;
  }
  __syncthreads();
  // pack per-batch A-fragments: [b][mt][ks][lane][i*2+slot]
  for (int idx = t; idx < BATCH * 4096; idx += 256) {
    int bb = idx >> 12;
    int r = idx & 4095;
    int mt = r >> 10, r2 = r & 1023, ks = r2 >> 9, r3 = r2 & 511;
    int lane = r3 >> 4, q = r3 & 15;
    int half = lane >> 4, m = mt * 16 + (lane & 15);
    int kk = ks * 32 + a_kmap(q >> 1, half, q & 1);
    frag[idx] = f2bf(wcomb[((size_t)bb * 64 + m) * 64 + kk]);
  }
}

// --- K9: grouped dwconv (reflect pad) + SiLU gate + BN, both halves fused --
static __device__ __forceinline__ int refl(int i) {
  if (i < 0) return -i;
  if (i >= IMG) return 2 * IMG - 2 - i;
  return i;
}

__global__ __launch_bounds__(256) void k_mlpdw(const float* __restrict__ m1, // [B][256][N]
                                               const float* __restrict__ w,  // [128][4][9]
                                               const float* __restrict__ g,
                                               const float* __restrict__ bb_,
                                               const float* __restrict__ mn,
                                               const float* __restrict__ vr,
                                               float* __restrict__ sout) {   // [B][64][N]
  size_t idx = (size_t)blockIdx.x * blockDim.x + threadIdx.x; // B*64*N
  int n = (int)(idx & 65535);
  int c = (int)((idx >> 16) & 63);
  int b = (int)(idx >> 22);
  int y = n >> 8, x = n & 255;
  int grp = c >> 1;
  const float* in1 = m1 + (((size_t)b * 256 + 4 * grp) << 16);
  const float* in2 = m1 + (((size_t)b * 256 + 128 + 4 * grp) << 16);
  const float* w1 = w + (size_t)c * 36;
  const float* w2 = w + (size_t)(64 + c) * 36;
  float a1 = 0.f, a2 = 0.f;
#pragma unroll
  for (int dy = 0; dy < 3; ++dy) {
    int yy = refl(y + dy - 1);
#pragma unroll
    for (int dx = 0; dx < 3; ++dx) {
      int x2 = refl(x + dx - 1);
      size_t off = (size_t)yy * IMG + x2;
#pragma unroll
      for (int j = 0; j < 4; ++j) {
        a1 += w1[j * 9 + dy * 3 + dx] * in1[((size_t)j << 16) + off];
        a2 += w2[j * 9 + dy * 3 + dx] * in2[((size_t)j << 16) + off];
      }
    }
  }
  float val = a1 * (1.0f / (1.0f + __expf(-a1))) * a2;   // silu(x1)*x2
  val = (val - mn[c]) * rsqrtf(vr[c] + 1e-5f) * g[c] + bb_[c];
  sout[idx] = val;
}

// ---------------------------------------------------------------------------
extern "C" void kernel_launch(void* const* d_in, const int* in_sizes, int n_in,
                              void* d_out, int out_size, void* d_ws, size_t ws_size,
                              hipStream_t stream) {
  const float* x        = (const float*)d_in[0];
  const float* ln1_w    = (const float*)d_in[1];
  const float* ln1_b    = (const float*)d_in[2];
  const float* ln2_w    = (const float*)d_in[3];
  const float* ln2_b    = (const float*)d_in[4];
  const float* dw_w     = (const float*)d_in[5];
  const float* q_w      = (const float*)d_in[6];
  const float* k_w      = (const float*)d_in[7];
  const float* v_w      = (const float*)d_in[8];
  const float* proj_w   = (const float*)d_in[9];
  const float* mlp_in_w = (const float*)d_in[10];
  const float* mlp_dw_w = (const float*)d_in[11];
  const float* bn_g     = (const float*)d_in[12];
  const float* bn_b     = (const float*)d_in[13];
  const float* bn_m     = (const float*)d_in[14];
  const float* bn_v     = (const float*)d_in[15];
  const float* out_w    = (const float*)d_in[16];
  float* outp = (float*)d_out;
  char* ws = (char*)d_ws;

  float* HN    = (float*)(ws + HN_OFF);
  float* HH    = (float*)(ws + HH_OFF);
  float* Q     = (float*)(ws + Q_OFF);
  float* V     = (float*)(ws + V_OFF);
  float* M1    = (float*)(ws + M1_OFF);
  float* SSQ   = (float*)(ws + SSQ_OFF);
  float* GRAM  = (float*)(ws + GRAM_OFF);
  float* WCOMB = (float*)(ws + WCOMB_OFF);
  unsigned short* FRQ = (unsigned short*)(ws + FRQ_OFF);
  unsigned short* FRM = (unsigned short*)(ws + FRM_OFF);
  unsigned short* FRO = (unsigned short*)(ws + FRO_OFF);
  unsigned short* FRW = (unsigned short*)(ws + FRW_OFF);

  // K0: zero accumulators (ssq 768 + gram 2048)
  k_zero<<<(2816 + 255) / 256, 256, 0, stream>>>(SSQ, 2816);
  // Kp: pack static weight fragments
  k_prep<<<16, 256, 0, stream>>>(q_w, 4, FRQ + 0);
  k_prep<<<16, 256, 0, stream>>>(k_w, 4, FRQ + 4096);
  k_prep<<<16, 256, 0, stream>>>(v_w, 4, FRQ + 8192);
  k_prep<<<64, 256, 0, stream>>>(mlp_in_w, 16, FRM);
  k_prep<<<16, 256, 0, stream>>>(out_w, 4, FRO);

  // K1: LN1
  k_ln<<<1024, 256, 0, stream>>>(x, ln1_w, ln1_b, HN);
  // K2: depthwise 3x3
  k_dw<<<65536, 256, 0, stream>>>(HN, dw_w, HH);
  // K3: fused q,k,v GEMMs (WMMA)
  k_wmma<3, 4, false, false><<<1024, 256, 0, stream>>>(HH, FRQ, Q, nullptr, T_EL);
  // K4: norms + Gram
  k_stats<<<dim3(32, 32), 256, 0, stream>>>(Q, Q + T_EL, V, SSQ, GRAM);
  // K5: softmax + fold proj -> per-batch Wcomb fragments
  k_attn<<<1, 256, 0, stream>>>(SSQ, GRAM, proj_w, WCOMB, FRW);
  // K6: d_out = x + Wcomb[b] * V   (attention + proj + residual, WMMA)
  k_wmma<1, 4, true, true><<<1024, 256, 0, stream>>>(V, FRW, outp, x, 0);
  // K7: LN2
  k_ln<<<1024, 256, 0, stream>>>(outp, ln2_w, ln2_b, HN);
  // K8: M1 = mlp_in_w * M0  (256 out channels, WMMA)
  k_wmma<1, 16, false, false><<<4096, 256, 0, stream>>>(HN, FRM, M1, nullptr, 0);
  // K9: grouped dwconv + SiLU gate + BN
  k_mlpdw<<<65536, 256, 0, stream>>>(M1, mlp_dw_w, bn_g, bn_b, bn_m, bn_v, HN);
  // K11: d_out += out_w * S  (WMMA, in-place residual)
  k_wmma<1, 4, true, false><<<1024, 256, 0, stream>>>(HN, FRO, outp, outp, 0);

  (void)in_sizes; (void)n_in; (void)out_size; (void)ws_size;
}
